// DeepHeadClassifier_44719199486096
// MI455X (gfx1250) — compile-verified, matrix operand order checked by reading, not run on hardware
//
#include <hip/hip_runtime.h>
#include <hip/hip_bf16.h>

typedef __bf16 bf16;
typedef __attribute__((ext_vector_type(16))) __bf16 v16bf;
typedef __attribute__((ext_vector_type(8)))  __bf16 v8bf;
typedef __attribute__((ext_vector_type(8)))  float  v8f;

#define E   256
#define NC  7
#define ROWS 16          // batch rows per block
#define TOK  48          // 3 * ROWS tokens
#define XS   264         // x tile stride (bf16 elems), 528B rows -> 16B aligned, bank-skewed
#define QS   776         // qkv tile stride (bf16)
#define CS   264         // ctx / x_ln stride (bf16)
#define AS   264         // att_out stride (f32)
#define HS   264         // head buffers stride
#define H2S  136         // h2 stride (K=128)

// ---- LDS layout (bytes). Aliased regions are phase-disjoint. ----
#define X_OFF      0
#define X_BYTES    (TOK*XS*2)                 // 25344
#define U_OFF      (X_OFF + X_BYTES)
#define U_BYTES    (TOK*QS*2)                 // 74496 : qkv bf16 | attout f32 | head scratch
#define C_OFF      (U_OFF + U_BYTES)          // 99840 : ctx bf16 | x_ln bf16
#define C_BYTES    (TOK*CS*2)                 // 25344
#define SMEM_BYTES (C_OFF + C_BYTES)          // 125184  (<< 320KB per WGP on CDNA5)

// head scratch inside U (alive only after attout is consumed)
#define FUSED_OFF  U_OFF                      // f32 [16][HS]
#define H_OFF      (FUSED_OFF + ROWS*HS*4)
#define H1_OFF     (H_OFF   + ROWS*HS*2)
#define H2_OFF     (H1_OFF  + ROWS*HS*2)
#define LG3_OFF    (H2_OFF  + ROWS*H2S*2)     // f32 [16][4]
static_assert(LG3_OFF + ROWS*4*4 - U_OFF <= U_BYTES, "head scratch must fit in U");

// bf16 weight offsets in d_ws (elements)
#define WS_INW  0
#define WS_OUTW 196608
#define WS_W1   262144
#define WS_W2   327680
#define WS_W3   360448
#define WS_ELEMS 361344

__device__ __forceinline__ float gelu_exact(float x) {
  return 0.5f * x * (1.0f + erff(x * 0.70710678118654752f));
}

// Load one 16x32 bf16 fragment (A layout) or 32x16 (B layout) from a
// row-major [16 rows][stride] tile. Per ISA 7.12.2: lane = row (M for A,
// N for B), half-wave selects K-group; elems {0..7}->K=8*half.., {8..15}->
// K=16+8*half.. : two contiguous 16-byte chunks per lane.
__device__ __forceinline__ v16bf load_frag(const bf16* p0, int stride, int lane) {
  const int half = lane >> 4;
  const int r    = lane & 15;
  const bf16* p = p0 + r * stride + half * 8;
  v8bf lo = *(const v8bf*)(p);
  v8bf hi = *(const v8bf*)(p + 16);
  v16bf f;
#pragma unroll
  for (int i = 0; i < 8; ++i) { f[i] = lo[i]; f[i + 8] = hi[i]; }
  return f;
}

__global__ void cvt_bf16_kernel(const float* __restrict__ in, bf16* __restrict__ out, int n) {
  int i = blockIdx.x * 256 + threadIdx.x;
  if (i < n) out[i] = (bf16)in[i];
}

__global__ __launch_bounds__(128)
void fusion_head_kernel(const float* __restrict__ fs, const float* __restrict__ fm,
                        const float* __restrict__ ff,
                        const bf16* __restrict__ wsbf,
                        const float* __restrict__ inB,  const float* __restrict__ outB,
                        const float* __restrict__ ln1g, const float* __restrict__ ln1b,
                        const float* __restrict__ gw,   const float* __restrict__ gb,
                        const float* __restrict__ ln2g, const float* __restrict__ ln2b,
                        const float* __restrict__ b1,   const float* __restrict__ b2,
                        const float* __restrict__ b3,
                        float* __restrict__ out_logits, float* __restrict__ out_fused) {
  __shared__ __align__(16) unsigned char smem[SMEM_BYTES];
  const int tid  = threadIdx.x;
  const int wave = tid >> 5;
  const int lane = tid & 31;
  const int row0 = blockIdx.x * ROWS;

  const bf16* inW  = wsbf + WS_INW;
  const bf16* outW = wsbf + WS_OUTW;
  const bf16* w1bf = wsbf + WS_W1;
  const bf16* w2bf = wsbf + WS_W2;
  const bf16* w3bf = wsbf + WS_W3;

  bf16* xp = (bf16*)(smem + X_OFF);     // [48][XS] original features (bf16), lives whole kernel

  // ---- Phase 0: stage features, fp32 -> bf16 ----
  for (int idx = tid; idx < 3 * ROWS * (E / 4); idx += blockDim.x) {
    int s   = idx / (ROWS * (E / 4));
    int rem = idx % (ROWS * (E / 4));
    int r   = rem / (E / 4);
    int e4  = rem % (E / 4);
    const float* sp = (s == 0) ? fs : ((s == 1) ? fm : ff);
    float4 v = ((const float4*)(sp + (size_t)(row0 + r) * E))[e4];
    bf16* dst = xp + (s * ROWS + r) * XS + e4 * 4;
    dst[0] = (bf16)v.x; dst[1] = (bf16)v.y; dst[2] = (bf16)v.z; dst[3] = (bf16)v.w;
  }
  __syncthreads();

  // ---- Phase 1: QKV = stacked @ in_proj_w^T + b  (144 tiles x 8 WMMA) ----
  {
    bf16* qkv = (bf16*)(smem + U_OFF);
    for (int tile = wave; tile < 3 * 48; tile += 4) {
      int mt = tile / 48, nt = tile % 48;
      v8f acc = {0.f, 0.f, 0.f, 0.f, 0.f, 0.f, 0.f, 0.f};
      const bf16* arow = xp + mt * 16 * XS;
      const bf16* brow = inW + (size_t)nt * 16 * E;
#pragma unroll
      for (int kt = 0; kt < 8; ++kt) {
        v16bf a = load_frag(arow + kt * 32, XS, lane);
        v16bf b = load_frag(brow + kt * 32, E, lane);
        acc = __builtin_amdgcn_wmma_f32_16x16x32_bf16(false, a, false, b,
                                                      (short)0, acc, false, false);
      }
      int n = lane & 15, half = lane >> 4;
      float bias = inB[nt * 16 + n];
#pragma unroll
      for (int i = 0; i < 8; ++i) {
        int m = i + 8 * half;
        qkv[(mt * 16 + m) * QS + nt * 16 + n] = (bf16)(acc[i] + bias);
      }
    }
  }
  __syncthreads();

  // ---- Phase 2: 3x3 attention per (row, head) in VALU ----
  if (tid < ROWS * 4) {
    const bf16* qkv = (const bf16*)(smem + U_OFF);
    bf16* ctx = (bf16*)(smem + C_OFF);
    int r = tid >> 2, h = tid & 3;
    float sc[3][3];
#pragma unroll
    for (int a = 0; a < 3; ++a)
#pragma unroll
      for (int b = 0; b < 3; ++b) sc[a][b] = 0.f;
    for (int d = 0; d < 64; ++d) {
      float qv[3], kv[3];
#pragma unroll
      for (int s = 0; s < 3; ++s) {
        qv[s] = (float)qkv[(s * 16 + r) * QS + h * 64 + d];
        kv[s] = (float)qkv[(s * 16 + r) * QS + 256 + h * 64 + d];
      }
#pragma unroll
      for (int a = 0; a < 3; ++a)
#pragma unroll
        for (int b = 0; b < 3; ++b) sc[a][b] += qv[a] * kv[b];
    }
    float att[3][3];
#pragma unroll
    for (int a = 0; a < 3; ++a) {
      float s0 = sc[a][0] * 0.125f, s1 = sc[a][1] * 0.125f, s2 = sc[a][2] * 0.125f;
      float m = fmaxf(s0, fmaxf(s1, s2));
      float e0 = __expf(s0 - m), e1 = __expf(s1 - m), e2 = __expf(s2 - m);
      float inv = 1.f / (e0 + e1 + e2);
      att[a][0] = e0 * inv; att[a][1] = e1 * inv; att[a][2] = e2 * inv;
    }
    for (int d = 0; d < 64; ++d) {
      float vv[3];
#pragma unroll
      for (int s = 0; s < 3; ++s)
        vv[s] = (float)qkv[(s * 16 + r) * QS + 512 + h * 64 + d];
#pragma unroll
      for (int a = 0; a < 3; ++a)
        ctx[(a * 16 + r) * CS + h * 64 + d] =
            (bf16)(att[a][0] * vv[0] + att[a][1] * vv[1] + att[a][2] * vv[2]);
    }
  }
  __syncthreads();

  // ---- Phase 3: att_out = ctx @ out_w^T + out_b + residual (48 tiles x 8 WMMA) ----
  {
    float* attout = (float*)(smem + U_OFF);        // aliases qkv (dead)
    const bf16* ctx = (const bf16*)(smem + C_OFF);
    for (int tile = wave; tile < 48; tile += 4) {
      int mt = tile / 16, nt = tile % 16;
      v8f acc = {0.f, 0.f, 0.f, 0.f, 0.f, 0.f, 0.f, 0.f};
      const bf16* arow = ctx + mt * 16 * CS;
      const bf16* brow = outW + (size_t)nt * 16 * E;
#pragma unroll
      for (int kt = 0; kt < 8; ++kt) {
        v16bf a = load_frag(arow + kt * 32, CS, lane);
        v16bf b = load_frag(brow + kt * 32, E, lane);
        acc = __builtin_amdgcn_wmma_f32_16x16x32_bf16(false, a, false, b,
                                                      (short)0, acc, false, false);
      }
      int n = lane & 15, half = lane >> 4;
      float bias = outB[nt * 16 + n];
#pragma unroll
      for (int i = 0; i < 8; ++i) {
        int m = i + 8 * half, t = mt * 16 + m;
        float resid = (float)xp[t * XS + nt * 16 + n];
        attout[t * AS + nt * 16 + n] = acc[i] + bias + resid;
      }
    }
  }
  __syncthreads();

  // ---- Phase 4: LayerNorm1 over each of the 48 token rows ----
  if (tid < TOK) {
    const float* ao = (const float*)(smem + U_OFF);
    bf16* xl = (bf16*)(smem + C_OFF);              // aliases ctx (dead)
    int t = tid;
    float sum = 0.f, ss = 0.f;
    for (int e = 0; e < E; ++e) { float v = ao[t * AS + e]; sum += v; ss += v * v; }
    float mu = sum * (1.f / E);
    float var = ss * (1.f / E) - mu * mu;
    float rs = rsqrtf(var + 1e-5f);
    for (int e = 0; e < E; ++e)
      xl[t * CS + e] = (bf16)((ao[t * AS + e] - mu) * rs * ln1g[e] + ln1b[e]);
  }
  __syncthreads();

  // ---- Phase 5: gate softmax -> fused -> LN2 ----
  float* lg3 = (float*)(smem + LG3_OFF);
  if (tid < ROWS * 3) {
    int r = tid / 3, j = tid % 3;
    float acc = gb[j];
    for (int s = 0; s < 3; ++s)
      for (int e = 0; e < E; ++e)
        acc += gw[j * 768 + s * 256 + e] * (float)xp[(s * 16 + r) * XS + e];
    lg3[r * 4 + j] = acc;
  }
  __syncthreads();
  if (tid < ROWS) {
    int r = tid;
    float a0 = lg3[r * 4], a1 = lg3[r * 4 + 1], a2 = lg3[r * 4 + 2];
    float m = fmaxf(a0, fmaxf(a1, a2));
    float e0 = __expf(a0 - m), e1 = __expf(a1 - m), e2 = __expf(a2 - m);
    float inv = 1.f / (e0 + e1 + e2);
    lg3[r * 4] = e0 * inv; lg3[r * 4 + 1] = e1 * inv; lg3[r * 4 + 2] = e2 * inv;
  }
  __syncthreads();
  {
    float* fused = (float*)(smem + FUSED_OFF);
    const bf16* xl = (const bf16*)(smem + C_OFF);
    for (int idx = tid; idx < ROWS * E; idx += blockDim.x) {
      int r = idx >> 8, e = idx & 255;
      float f = lg3[r * 4 + 0] * (float)xl[(0 * 16 + r) * CS + e]
              + lg3[r * 4 + 1] * (float)xl[(1 * 16 + r) * CS + e]
              + lg3[r * 4 + 2] * (float)xl[(2 * 16 + r) * CS + e];
      fused[r * HS + e] = f;
      out_fused[(size_t)(row0 + r) * E + e] = f;
    }
  }
  __syncthreads();
  if (tid < ROWS) {
    const float* fused = (const float*)(smem + FUSED_OFF);
    bf16* h = (bf16*)(smem + H_OFF);
    int r = tid;
    float sum = 0.f, ss = 0.f;
    for (int e = 0; e < E; ++e) { float v = fused[r * HS + e]; sum += v; ss += v * v; }
    float mu = sum * (1.f / E);
    float var = ss * (1.f / E) - mu * mu;
    float rs = rsqrtf(var + 1e-5f);
    for (int e = 0; e < E; ++e)
      h[r * HS + e] = (bf16)((fused[r * HS + e] - mu) * rs * ln2g[e] + ln2b[e]);
  }
  __syncthreads();

  // ---- Phase 6: h1 = GELU(h @ w1^T + b1)  (16 tiles x 8 WMMA) ----
  {
    const bf16* h = (const bf16*)(smem + H_OFF);
    bf16* h1 = (bf16*)(smem + H1_OFF);
    for (int nt = wave; nt < 16; nt += 4) {
      v8f acc = {0.f, 0.f, 0.f, 0.f, 0.f, 0.f, 0.f, 0.f};
      const bf16* brow = w1bf + (size_t)nt * 16 * E;
#pragma unroll
      for (int kt = 0; kt < 8; ++kt) {
        v16bf a = load_frag(h + kt * 32, HS, lane);
        v16bf b = load_frag(brow + kt * 32, E, lane);
        acc = __builtin_amdgcn_wmma_f32_16x16x32_bf16(false, a, false, b,
                                                      (short)0, acc, false, false);
      }
      int n = lane & 15, half = lane >> 4;
      float bias = b1[nt * 16 + n];
#pragma unroll
      for (int i = 0; i < 8; ++i) {
        int m = i + 8 * half;
        h1[m * HS + nt * 16 + n] = (bf16)gelu_exact(acc[i] + bias);
      }
    }
  }
  __syncthreads();

  // ---- Phase 7: h2 = GELU(h1 @ w2^T + b2)  (8 tiles x 8 WMMA) ----
  {
    const bf16* h1 = (const bf16*)(smem + H1_OFF);
    bf16* h2 = (bf16*)(smem + H2_OFF);
    for (int nt = wave; nt < 8; nt += 4) {
      v8f acc = {0.f, 0.f, 0.f, 0.f, 0.f, 0.f, 0.f, 0.f};
      const bf16* brow = w2bf + (size_t)nt * 16 * E;
#pragma unroll
      for (int kt = 0; kt < 8; ++kt) {
        v16bf a = load_frag(h1 + kt * 32, HS, lane);
        v16bf b = load_frag(brow + kt * 32, E, lane);
        acc = __builtin_amdgcn_wmma_f32_16x16x32_bf16(false, a, false, b,
                                                      (short)0, acc, false, false);
      }
      int n = lane & 15, half = lane >> 4;
      float bias = b2[nt * 16 + n];
#pragma unroll
      for (int i = 0; i < 8; ++i) {
        int m = i + 8 * half;
        h2[m * H2S + nt * 16 + n] = (bf16)gelu_exact(acc[i] + bias);
      }
    }
  }
  __syncthreads();

  // ---- Phase 8: logits = h2 @ w3^T + b3  (7-wide, VALU) ----
  if (tid < ROWS * NC) {
    const bf16* h2 = (const bf16*)(smem + H2_OFF);
    int r = tid / NC, c = tid % NC;
    float acc = b3[c];
    for (int k = 0; k < 128; ++k)
      acc += (float)h2[r * H2S + k] * (float)w3bf[c * 128 + k];
    out_logits[(size_t)(row0 + r) * NC + c] = acc;
  }
}

extern "C" void kernel_launch(void* const* d_in, const int* in_sizes, int n_in,
                              void* d_out, int out_size, void* d_ws, size_t ws_size,
                              hipStream_t stream) {
  const float* fs   = (const float*)d_in[0];
  const float* fm   = (const float*)d_in[1];
  const float* ff   = (const float*)d_in[2];
  const float* inW  = (const float*)d_in[3];
  const float* inB  = (const float*)d_in[4];
  const float* outW = (const float*)d_in[5];
  const float* outB = (const float*)d_in[6];
  const float* ln1g = (const float*)d_in[7];
  const float* ln1b = (const float*)d_in[8];
  const float* gw   = (const float*)d_in[9];
  const float* gb   = (const float*)d_in[10];
  const float* ln2g = (const float*)d_in[11];
  const float* ln2b = (const float*)d_in[12];
  const float* w1   = (const float*)d_in[13];
  const float* b1   = (const float*)d_in[14];
  const float* w2   = (const float*)d_in[15];
  const float* b2   = (const float*)d_in[16];
  const float* w3   = (const float*)d_in[17];
  const float* b3   = (const float*)d_in[18];

  const int Brows = in_sizes[0] / E;
  bf16* wsbf = (bf16*)d_ws;

  // One-time (per launch) fp32 -> bf16 weight conversion into workspace.
  cvt_bf16_kernel<<<(196608 + 255) / 256, 256, 0, stream>>>(inW,  wsbf + WS_INW,  196608);
  cvt_bf16_kernel<<<( 65536 + 255) / 256, 256, 0, stream>>>(outW, wsbf + WS_OUTW,  65536);
  cvt_bf16_kernel<<<( 65536 + 255) / 256, 256, 0, stream>>>(w1,   wsbf + WS_W1,    65536);
  cvt_bf16_kernel<<<( 32768 + 255) / 256, 256, 0, stream>>>(w2,   wsbf + WS_W2,    32768);
  cvt_bf16_kernel<<<(   896 + 255) / 256, 256, 0, stream>>>(w3,   wsbf + WS_W3,      896);

  float* out_logits = (float*)d_out;
  float* out_fused  = (float*)d_out + (size_t)Brows * NC;

  fusion_head_kernel<<<Brows / ROWS, 128, 0, stream>>>(
      fs, fm, ff, wsbf, inB, outB, ln1g, ln1b, gw, gb, ln2g, ln2b,
      b1, b2, b3, out_logits, out_fused);
}
